// UnifiedTransformerBlock_64209761075862
// MI455X (gfx1250) — matrix operand order repaired and checked by reading, not compiled, not run on hardware
//
#include <hip/hip_runtime.h>
#include <hip/hip_bf16.h>
#include <math.h>

// ---------------------------------------------------------------------------
// MI455X (gfx1250) transformer block: bf16 WMMA GEMMs + routed MoE.
// wave32; 256-thread (8-wave) workgroups computing 128x64 output tiles.
// Ping-pong double-buffered LDS; GLOBAL_LOAD_ASYNC_TO_LDS_B128 stages the
// next A tile and the next B tile's global load is issued into registers
// BEFORE the WMMA block (scatter to LDS after), so all global latency hides
// under the matrix ops. One s_wait_asynccnt + one barrier per k-step.
// ---------------------------------------------------------------------------

typedef __bf16 bf16_t;
typedef __attribute__((ext_vector_type(16))) __bf16 v16bf;
typedef __attribute__((ext_vector_type(8)))  __bf16 v8bf;
typedef __attribute__((ext_vector_type(8)))  float  v8f;
typedef int v4i_vs __attribute__((vector_size(16)));   // matches builtin proto

#define DD   1024   // model dim
#define TT   1024   // seq len
#define BB   2      // batch
#define NT   2048   // B*T tokens
#define NHH  16     // heads
#define DHH  64     // head dim
#define FFN2 2048   // FFN hidden
#define EE   8      // experts
#define CAP  2048   // per-expert token capacity

// ---- async global->LDS staging (gfx1250 ASYNCcnt path) --------------------
#if __has_builtin(__builtin_amdgcn_global_load_async_to_lds_b128)
__device__ __forceinline__ void async_b128(const void* g, void* l) {
  __builtin_amdgcn_global_load_async_to_lds_b128((v4i_vs*)g, (v4i_vs*)l, 0, 0);
}
__device__ __forceinline__ void async_wait() {
  asm volatile("s_wait_asynccnt 0" ::: "memory");
}
#else
__device__ __forceinline__ void async_b128(const void* g, void* l) {
  *(v8bf*)l = *(const v8bf*)g;
}
__device__ __forceinline__ void async_wait() {}
#endif

__device__ __forceinline__ v8f wmma_bf16f32(v16bf a, v16bf b, v8f c) {
  // v_wmma_f32_16x16x32_bf16: D = A(16x32) * B(32x16) + C(16x16 f32)
  return __builtin_amdgcn_wmma_f32_16x16x32_bf16(false, a, false, b,
                                                 (short)0, c, false, false);
}

__device__ __forceinline__ v16bf cat8(v8bf lo, v8bf hi) {
  return __builtin_shufflevector(lo, hi, 0, 1, 2, 3, 4, 5, 6, 7,
                                 8, 9, 10, 11, 12, 13, 14, 15);
}

// A-matrix 16x32 bf16 fragment: lane<16 holds row m=lane,
// K 0..15; lane>=16 holds K 16..31.
__device__ __forceinline__ void load_a16x32(const bf16_t* src, int ld, int lane,
                                            v16bf& a) {
  const int m  = lane & 15;
  const int kb = (lane >> 4) << 4;
  const bf16_t* p = src + (size_t)m * ld + kb;
  v8bf lo = *(const v8bf*)(p);
  v8bf hi = *(const v8bf*)(p + 8);
  a = cat8(lo, hi);
}

// B fragment from fragment-major swizzled LDS: lane's 16 values contiguous.
__device__ __forceinline__ void load_b_swz(const bf16_t* frag_base, int lane,
                                           v16bf& b) {
  const v8bf* p = (const v8bf*)(frag_base + lane * 16);
  b = cat8(p[0], p[1]);
}

// Split B-tile staging: issue the coalesced b128 global load (256-thread
// version: one vector per thread) ...
__device__ __forceinline__ v8bf loadv_b_256(const bf16_t* __restrict__ Bg,
                                            size_t ldb, int tid) {
  int r = tid >> 3, c8 = (tid & 7) * 8;
  return *(const v8bf*)(Bg + (size_t)r * ldb + c8);
}
// ... and scatter into fragment-major LDS BsF[j=N/16][lane][i=K%16] later.
__device__ __forceinline__ void scatter_b_256(v8bf d,
                                              bf16_t (*__restrict__ BsF)[32][16],
                                              int tid) {
  int r = tid >> 3, c8 = (tid & 7) * 8;
  int j = c8 >> 4;
  int lane0 = ((r >> 4) << 4) + (c8 & 15);
  int i = r & 15;
#pragma unroll
  for (int u = 0; u < 8; ++u) BsF[j][lane0 + u][i] = d[u];
}

// 128-thread split versions (attention): two vectors per thread.
__device__ __forceinline__ void loadv_b_128(const bf16_t* __restrict__ Bg,
                                            size_t ldb, int tid, v8bf d[2]) {
#pragma unroll
  for (int it = 0; it < 2; ++it) {
    int v = tid + it * 128;
    int r = v >> 3, c8 = (v & 7) * 8;
    d[it] = *(const v8bf*)(Bg + (size_t)r * ldb + c8);
  }
}
__device__ __forceinline__ void scatter_b_128(const v8bf d[2],
                                              bf16_t (*__restrict__ BsF)[32][16],
                                              int tid) {
#pragma unroll
  for (int it = 0; it < 2; ++it) {
    int v = tid + it * 128;
    int r = v >> 3, c8 = (v & 7) * 8;
    int j = c8 >> 4;
    int lane0 = ((r >> 4) << 4) + (c8 & 15);
    int i = r & 15;
#pragma unroll
    for (int u = 0; u < 8; ++u) BsF[j][lane0 + u][i] = d[it][u];
  }
}

// Stage a 128(M) x 32(K) row-major A tile via async b128 (256 threads).
__device__ __forceinline__ void stage_a_async_128(const bf16_t* __restrict__ Ag,
                                                  size_t lda,
                                                  bf16_t (*__restrict__ As)[32],
                                                  int tid) {
#pragma unroll
  for (int it = 0; it < 2; ++it) {
    int v = tid + it * 256;
    int r = v >> 2, c8 = (v & 3) * 8;
    async_b128(Ag + (size_t)r * lda + c8, &As[r][c8]);
  }
}

// ---------------------------------------------------------------------------
// fp32 -> bf16 streaming convert
// ---------------------------------------------------------------------------
__global__ void __launch_bounds__(256) cvt_kernel(const float* __restrict__ in,
                                                  bf16_t* __restrict__ out, int n) {
  for (int i = blockIdx.x * 256 + threadIdx.x; i < n; i += gridDim.x * 256)
    out[i] = (bf16_t)in[i];
}

// ---------------------------------------------------------------------------
// Row LayerNorm (fp32 in, bf16 out), one 256-thread block per row
// ---------------------------------------------------------------------------
__global__ void __launch_bounds__(256) ln_kernel(const float* __restrict__ x,
                                                 const float* __restrict__ g,
                                                 const float* __restrict__ b,
                                                 bf16_t* __restrict__ out) {
  __shared__ float ssum[256], ssq[256];
  const int row = blockIdx.x, tid = threadIdx.x;
  const float* xr = x + (size_t)row * DD;
  float s = 0.f, q = 0.f;
  for (int c = tid; c < DD; c += 256) { float v = xr[c]; s += v; q += v * v; }
  ssum[tid] = s; ssq[tid] = q; __syncthreads();
  for (int off = 128; off > 0; off >>= 1) {
    if (tid < off) { ssum[tid] += ssum[tid + off]; ssq[tid] += ssq[tid + off]; }
    __syncthreads();
  }
  const float mu  = ssum[0] * (1.f / DD);
  const float var = ssq[0] * (1.f / DD) - mu * mu;
  const float inv = rsqrtf(var + 1e-5f);
  for (int c = tid; c < DD; c += 256)
    out[(size_t)row * DD + c] = (bf16_t)((xr[c] - mu) * inv * g[c] + b[c]);
}

// ---------------------------------------------------------------------------
// Generic bf16 WMMA GEMM: C[M,N] = A[M,K] * B[K,N] (+res), optional 2nd dst.
// Grid: (M/128, N/64); block: 256 threads = 8 waves; ping-pong pipelined.
// ---------------------------------------------------------------------------
__global__ void __launch_bounds__(256) gemm_bf16_kernel(
    const bf16_t* __restrict__ A, int lda,
    const bf16_t* __restrict__ Bw, int ldb,
    float* __restrict__ C, int ldc, int K,
    const float* __restrict__ res, float* __restrict__ C2) {
  __shared__ bf16_t As[2][128][32];
  __shared__ bf16_t BsF[2][4][32][16];
  const int m0 = blockIdx.x * 128, n0 = blockIdx.y * 64;
  const int tid = threadIdx.x, wave = tid >> 5, lane = tid & 31;
  v8f acc[4] = { {}, {}, {}, {} };
  // prologue: stage k-tile 0 into buffer 0
  stage_a_async_128(A + (size_t)m0 * lda, (size_t)lda, As[0], tid);
  scatter_b_256(loadv_b_256(Bw + n0, (size_t)ldb, tid), BsF[0], tid);
  async_wait();
  __syncthreads();
  int p = 0;
  for (int k0 = 0; k0 < K; k0 += 32) {
    const int q = p ^ 1;
    const bool more = (k0 + 32 < K);
    v8bf dB{};
    if (more) {   // issue next tile's loads before compute
      stage_a_async_128(A + (size_t)m0 * lda + k0 + 32, (size_t)lda, As[q], tid);
      dB = loadv_b_256(Bw + (size_t)(k0 + 32) * ldb + n0, (size_t)ldb, tid);
    }
    if (k0 + 64 < K)     // global_prefetch_b8 two tiles ahead
      __builtin_prefetch(&Bw[(size_t)(k0 + 64 + (tid >> 3)) * ldb + n0 + (tid & 7) * 8], 0, 1);
    v16bf a;
    load_a16x32(&As[p][wave * 16][0], 32, lane, a);
#pragma unroll
    for (int j = 0; j < 4; ++j) {
      v16bf b;
      load_b_swz(&BsF[p][j][0][0], lane, b);
      acc[j] = wmma_bf16f32(a, b, acc[j]);
    }
    if (more) scatter_b_256(dB, BsF[q], tid);   // wait_loadcnt lands here
    async_wait();
    __syncthreads();
    p = q;
  }
  const int n = lane & 15, mb = (lane >> 4) << 3;
#pragma unroll
  for (int j = 0; j < 4; ++j)
#pragma unroll
    for (int r = 0; r < 8; ++r) {
      size_t off = (size_t)(m0 + wave * 16 + mb + r) * ldc + n0 + j * 16 + n;
      float v = acc[j][r];
      if (res) v += res[off];
      C[off] = v;
      if (C2) C2[off] = v;
    }
}

// ---------------------------------------------------------------------------
// RoPE + transpose to [B,H,T,DH] bf16 buffers
// ---------------------------------------------------------------------------
__global__ void __launch_bounds__(256) rope_kernel(const float* __restrict__ qkv,
                                                   bf16_t* __restrict__ qb,
                                                   bf16_t* __restrict__ kbuf,
                                                   bf16_t* __restrict__ vbuf) {
  int gidx = blockIdx.x * 256 + threadIdx.x;
  if (gidx >= BB * TT * NHH * 32) return;
  int j  = gidx & 31;
  int h  = (gidx >> 5) & 15;
  int t  = (gidx >> 9) & 1023;
  int bi = gidx >> 19;
  size_t row = (size_t)bi * TT + t;
  const float* src = qkv + row * (3 * DD);
  float inv = __powf(10000.f, -(float)(2 * j) / (float)DHH);
  float ang = (float)t * inv;
  float cs = __cosf(ang), sn = __sinf(ang);
  size_t dst = ((size_t)(bi * NHH + h) * TT + t) * DHH + 2 * j;
  {
    float x1 = src[h * DHH + 2 * j], x2 = src[h * DHH + 2 * j + 1];
    qb[dst]     = (bf16_t)(x1 * cs - x2 * sn);
    qb[dst + 1] = (bf16_t)(x2 * cs + x1 * sn);
  }
  {
    float x1 = src[DD + h * DHH + 2 * j], x2 = src[DD + h * DHH + 2 * j + 1];
    kbuf[dst]     = (bf16_t)(x1 * cs - x2 * sn);
    kbuf[dst + 1] = (bf16_t)(x2 * cs + x1 * sn);
  }
  vbuf[dst]     = (bf16_t)src[2 * DD + h * DHH + 2 * j];
  vbuf[dst + 1] = (bf16_t)src[2 * DD + h * DHH + 2 * j + 1];
}

// ---------------------------------------------------------------------------
// Attention: one block (128 thr) per (b,h, 16-query tile). S=QK^T via WMMA
// into a 16x1024 LDS strip, fp32 softmax, O=P*V with pipelined V staging.
// ---------------------------------------------------------------------------
__global__ void __launch_bounds__(128) attn_kernel(const bf16_t* __restrict__ qb,
                                                   const bf16_t* __restrict__ kbv,
                                                   const bf16_t* __restrict__ vbv,
                                                   bf16_t* __restrict__ ob) {
  __shared__ float  S[16][1024];        // 64 KB score strip
  __shared__ bf16_t P[16][1024];        // 32 KB probabilities
  __shared__ bf16_t Qs[16][64];
  __shared__ bf16_t VsF[2][4][32][16];  // ping-pong swizzled V chunks
  __shared__ float  red[128], rowmax[16], rowsum[16];
  const int bh = blockIdx.x >> 6;
  const int qt = blockIdx.x & 63;
  const int bidx = bh >> 4, h = bh & 15;
  const int tid = threadIdx.x, wave = tid >> 5, lane = tid & 31;
  const bf16_t* kg = kbv + (size_t)bh * TT * DHH;
  const bf16_t* vg = vbv + (size_t)bh * TT * DHH;
  const bf16_t* qg = qb  + (size_t)bh * TT * DHH + (size_t)qt * 16 * DHH;
  async_b128(qg + tid * 8, ((bf16_t*)Qs) + tid * 8);   // 16x64 Q tile
  async_wait();
  __syncthreads();
  v16bf a0, a1;
  load_a16x32(&Qs[0][0],  64, lane, a0);
  load_a16x32(&Qs[0][32], 64, lane, a1);
  const int n = lane & 15, mb = (lane >> 4) << 3;
  // Phase 1: scores. K^T fragment data is contiguous per lane in [T,DH].
  const int kb2 = (lane >> 4) << 4;
  for (int kt = wave; kt < 64; kt += 4) {
    const bf16_t* krow = kg + (size_t)(kt * 16 + n) * DHH + kb2;
    const v8bf* kp = (const v8bf*)krow;
    v16bf b0 = cat8(kp[0], kp[1]);         // K columns kb2..kb2+15
    v16bf b1 = cat8(kp[2], kp[3]);         // K columns kb2+16..kb2+31
    v8f acc = {};
    acc = wmma_bf16f32(a0, b0, acc);
    acc = wmma_bf16f32(a1, b1, acc);
#pragma unroll
    for (int r = 0; r < 8; ++r) S[mb + r][kt * 16 + n] = acc[r] * 0.125f;
  }
  __syncthreads();
  // Phase 2: softmax over 1024 cols, 8 threads per row
  const int row = tid >> 3, sub = tid & 7;
  float mx = -1e30f;
  for (int c = sub; c < TT; c += 8) mx = fmaxf(mx, S[row][c]);
  red[tid] = mx; __syncthreads();
  if (sub == 0) {
    float mm = red[tid];
#pragma unroll
    for (int i = 1; i < 8; ++i) mm = fmaxf(mm, red[tid + i]);
    rowmax[row] = mm;
  }
  __syncthreads();
  float s = 0.f; const float rm = rowmax[row];
  for (int c = sub; c < TT; c += 8) { float e = __expf(S[row][c] - rm); S[row][c] = e; s += e; }
  red[tid] = s; __syncthreads();
  if (sub == 0) {
    float ss = 0.f;
#pragma unroll
    for (int i = 0; i < 8; ++i) ss += red[tid + i];
    rowsum[row] = ss;
  }
  __syncthreads();
  const float invs = 1.f / rowsum[row];
  for (int c = sub; c < TT; c += 8) P[row][c] = (bf16_t)(S[row][c] * invs);
  // Phase 3: O = P(16x1024) @ V(1024x64); pipelined V staging.
  const int nc0 = wave * 16;
  {
    v8bf d0[2];
    loadv_b_128(vg, (size_t)DHH, tid, d0);
    scatter_b_128(d0, VsF[0], tid);
  }
  __syncthreads();
  int p = 0;
  v8f acc = {};
  for (int kt = 0; kt < 32; ++kt) {
    const int q = p ^ 1;
    const bool more = (kt + 1 < 32);
    v8bf dV[2];
    if (more) loadv_b_128(vg + (size_t)(kt + 1) * 32 * DHH, (size_t)DHH, tid, dV);
    v16bf a; load_a16x32(&P[0][kt * 32], TT, lane, a);
    v16bf b; load_b_swz(&VsF[p][wave][0][0], lane, b);
    acc = wmma_bf16f32(a, b, acc);
    if (more) scatter_b_128(dV, VsF[q], tid);
    __syncthreads();
    p = q;
  }
#pragma unroll
  for (int r = 0; r < 8; ++r) {
    int trow = bidx * TT + qt * 16 + mb + r;
    ob[(size_t)trow * DD + h * DHH + nc0 + n] = (bf16_t)acc[r];
  }
}

// ---------------------------------------------------------------------------
// Gating: one wave per token; top-2 routing + LB-loss accumulation.
// ---------------------------------------------------------------------------
__global__ void __launch_bounds__(32) gate_kernel(const bf16_t* __restrict__ xn2,
                                                  const float* __restrict__ gw,
                                                  const float* __restrict__ ebias,
                                                  int* __restrict__ cnts,
                                                  int* __restrict__ lists,
                                                  float* __restrict__ gts,
                                                  float* __restrict__ probs_acc) {
  const int t = blockIdx.x;
  const int lane = threadIdx.x;
  float acc[EE];
#pragma unroll
  for (int e = 0; e < EE; ++e) acc[e] = 0.f;
  for (int d = lane; d < DD; d += 32) {
    float xv = (float)xn2[(size_t)t * DD + d];
#pragma unroll
    for (int e = 0; e < EE; ++e) acc[e] += xv * gw[d * EE + e];
  }
#pragma unroll
  for (int off = 16; off > 0; off >>= 1)
#pragma unroll
    for (int e = 0; e < EE; ++e) acc[e] += __shfl_down(acc[e], off, 32);
  if (lane == 0) {
    float sc[EE], mx = -1e30f;
#pragma unroll
    for (int e = 0; e < EE; ++e) { sc[e] = acc[e] + ebias[e]; mx = fmaxf(mx, sc[e]); }
    float se = 0.f, pe[EE];
#pragma unroll
    for (int e = 0; e < EE; ++e) { pe[e] = __expf(sc[e] - mx); se += pe[e]; }
    float rs = 1.f / se;
#pragma unroll
    for (int e = 0; e < EE; ++e) atomicAdd(&probs_acc[e], pe[e] * rs);
    int i0 = 0;
#pragma unroll
    for (int e = 1; e < EE; ++e) if (sc[e] > sc[i0]) i0 = e;
    int i1 = (i0 == 0) ? 1 : 0;
#pragma unroll
    for (int e = 0; e < EE; ++e) if (e != i0 && sc[e] > sc[i1]) i1 = e;
    float e1 = __expf(sc[i1] - sc[i0]);
    float g0 = 1.f / (1.f + e1), g1 = e1 / (1.f + e1);
    int p0 = atomicAdd(&cnts[i0], 1);
    lists[i0 * CAP + p0] = t; gts[i0 * CAP + p0] = g0;
    int p1 = atomicAdd(&cnts[i1], 1);
    lists[i1 * CAP + p1] = t; gts[i1 * CAP + p1] = g1;
  }
}

// ---------------------------------------------------------------------------
// MoE GEMM1: gathered tokens x w1[e], fused SwiGLU. Grid (CAP/128,FFN/64,E).
// ---------------------------------------------------------------------------
__global__ void __launch_bounds__(256) moe_gemm1_kernel(
    const bf16_t* __restrict__ xn2, const bf16_t* __restrict__ w1,
    const int* __restrict__ lists, const int* __restrict__ cnts,
    bf16_t* __restrict__ act) {
  __shared__ bf16_t As[2][128][32];
  __shared__ bf16_t BsF1[2][4][32][16];
  __shared__ bf16_t BsF2[2][4][32][16];
  const int e = blockIdx.z;
  const int cnt = cnts[e];
  const int m0 = blockIdx.x * 128;
  if (m0 >= cnt) return;
  const int n0 = blockIdx.y * 64;
  const int tid = threadIdx.x, wave = tid >> 5, lane = tid & 31;
  const int* lst = lists + e * CAP;
  const bf16_t* w1e = w1 + (size_t)e * DD * (2 * FFN2);
  v8f acc1[4] = { {}, {}, {}, {} };
  v8f acc2[4] = { {}, {}, {}, {} };
  auto stage_a = [&](int buf, int k0) {    // gathered async A staging
#pragma unroll
    for (int it = 0; it < 2; ++it) {
      int v = tid + it * 256;
      int r = v >> 2, c8 = (v & 3) * 8;
      int mr = m0 + r;
      int srow = lst[(mr < cnt) ? mr : 0];
      async_b128(xn2 + (size_t)srow * DD + k0 + c8, &As[buf][r][c8]);
    }
  };
  stage_a(0, 0);
  scatter_b_256(loadv_b_256(w1e + n0,        (size_t)(2 * FFN2), tid), BsF1[0], tid);
  scatter_b_256(loadv_b_256(w1e + FFN2 + n0, (size_t)(2 * FFN2), tid), BsF2[0], tid);
  async_wait();
  __syncthreads();
  int p = 0;
  for (int k0 = 0; k0 < DD; k0 += 32) {
    const int q = p ^ 1;
    const bool more = (k0 + 32 < DD);
    v8bf dB1{}, dB2{};
    if (more) {
      stage_a(q, k0 + 32);
      dB1 = loadv_b_256(w1e + (size_t)(k0 + 32) * (2 * FFN2) + n0,        (size_t)(2 * FFN2), tid);
      dB2 = loadv_b_256(w1e + (size_t)(k0 + 32) * (2 * FFN2) + FFN2 + n0, (size_t)(2 * FFN2), tid);
    }
    v16bf a; load_a16x32(&As[p][wave * 16][0], 32, lane, a);
#pragma unroll
    for (int j = 0; j < 4; ++j) {
      v16bf b;
      load_b_swz(&BsF1[p][j][0][0], lane, b);
      acc1[j] = wmma_bf16f32(a, b, acc1[j]);
      load_b_swz(&BsF2[p][j][0][0], lane, b);
      acc2[j] = wmma_bf16f32(a, b, acc2[j]);
    }
    if (more) { scatter_b_256(dB1, BsF1[q], tid); scatter_b_256(dB2, BsF2[q], tid); }
    async_wait();
    __syncthreads();
    p = q;
  }
  const int n = lane & 15, mb = (lane >> 4) << 3;
#pragma unroll
  for (int j = 0; j < 4; ++j)
#pragma unroll
    for (int r = 0; r < 8; ++r) {
      int mr = m0 + wave * 16 + mb + r;
      if (mr < cnt) {
        float h1 = acc1[j][r], h2 = acc2[j][r];
        float sw = h1 / (1.f + __expf(-h1));          // silu
        act[((size_t)e * CAP + mr) * FFN2 + n0 + j * 16 + n] = (bf16_t)(sw * h2);
      }
    }
}

// ---------------------------------------------------------------------------
// MoE GEMM2: act[e] x w2[e]; gate-weighted atomic scatter-add into d_out.
// Grid (CAP/128, D/64, E).
// ---------------------------------------------------------------------------
__global__ void __launch_bounds__(256) moe_gemm2_kernel(
    const bf16_t* __restrict__ act, const bf16_t* __restrict__ w2,
    const int* __restrict__ lists, const int* __restrict__ cnts,
    const float* __restrict__ gts, float* __restrict__ out) {
  __shared__ bf16_t As[2][128][32];
  __shared__ bf16_t BsF[2][4][32][16];
  const int e = blockIdx.z;
  const int cnt = cnts[e];
  const int m0 = blockIdx.x * 128;
  if (m0 >= cnt) return;
  const int n0 = blockIdx.y * 64;
  const int tid = threadIdx.x, wave = tid >> 5, lane = tid & 31;
  const bf16_t* ae  = act + (size_t)e * CAP * FFN2;
  const bf16_t* w2e = w2  + (size_t)e * FFN2 * DD;
  v8f acc[4] = { {}, {}, {}, {} };
  stage_a_async_128(ae + (size_t)m0 * FFN2, (size_t)FFN2, As[0], tid);
  scatter_b_256(loadv_b_256(w2e + n0, (size_t)DD, tid), BsF[0], tid);
  async_wait();
  __syncthreads();
  int p = 0;
  for (int k0 = 0; k0 < FFN2; k0 += 32) {
    const int q = p ^ 1;
    const bool more = (k0 + 32 < FFN2);
    v8bf dB{};
    if (more) {
      stage_a_async_128(ae + (size_t)m0 * FFN2 + k0 + 32, (size_t)FFN2, As[q], tid);
      dB = loadv_b_256(w2e + (size_t)(k0 + 32) * DD + n0, (size_t)DD, tid);
    }
    v16bf a; load_a16x32(&As[p][wave * 16][0], 32, lane, a);
#pragma unroll
    for (int j = 0; j < 4; ++j) {
      v16bf b; load_b_swz(&BsF[p][j][0][0], lane, b);
      acc[j] = wmma_bf16f32(a, b, acc[j]);
    }
    if (more) scatter_b_256(dB, BsF[q], tid);
    async_wait();
    __syncthreads();
    p = q;
  }
  const int n = lane & 15, mb = (lane >> 4) << 3;
  const int* lst = lists + e * CAP;
  const float* gte = gts + e * CAP;
#pragma unroll
  for (int j = 0; j < 4; ++j)
#pragma unroll
    for (int r = 0; r < 8; ++r) {
      int mr = m0 + wave * 16 + mb + r;
      if (mr < cnt) {
        int trow = lst[mr];
        atomicAdd(&out[(size_t)trow * DD + n0 + j * 16 + n], acc[j][r] * gte[mr]);
      }
    }
}

__global__ void zero_kernel(int* cnts, float* probs_acc) {
  int i = threadIdx.x;
  if (i < EE) { cnts[i] = 0; probs_acc[i] = 0.f; }
}

__global__ void loss_kernel(const int* __restrict__ cnts,
                            const float* __restrict__ probs_acc,
                            float* __restrict__ out_loss) {
  if (threadIdx.x == 0) {
    float usum = 0.f, psum = 0.f;
    for (int e = 0; e < EE; ++e) { usum += (float)cnts[e]; psum += probs_acc[e]; }
    float lb = 0.f;
    for (int e = 0; e < EE; ++e)
      lb += ((float)cnts[e] / usum) * (probs_acc[e] / psum);
    *out_loss = lb * (float)EE;
  }
}

// ---------------------------------------------------------------------------
extern "C" void kernel_launch(void* const* d_in, const int* in_sizes, int n_in,
                              void* d_out, int out_size, void* d_ws, size_t ws_size,
                              hipStream_t stream) {
  const float* x      = (const float*)d_in[0];
  const float* ln1_g  = (const float*)d_in[1];
  const float* ln1_b  = (const float*)d_in[2];
  const float* qkv_w  = (const float*)d_in[3];
  const float* out_w  = (const float*)d_in[4];
  const float* ln2_g  = (const float*)d_in[5];
  const float* ln2_b  = (const float*)d_in[6];
  const float* gate_w = (const float*)d_in[7];
  const float* ebias  = (const float*)d_in[8];
  const float* w1     = (const float*)d_in[9];
  const float* w2     = (const float*)d_in[10];
  (void)in_sizes; (void)n_in; (void)out_size; (void)ws_size;

  float* out_x    = (float*)d_out;                    // [B,T,D]
  float* out_loss = out_x + (size_t)BB * TT * DD;     // scalar

  char* base = (char*)d_ws;
  size_t off = 0;
  auto alloc = [&](size_t bytes) -> void* {
    void* p = base + off;
    off = (off + bytes + 255) & ~(size_t)255;
    return p;
  };
  bf16_t* qkv_w_bf = (bf16_t*)alloc((size_t)DD * 3 * DD * 2);
  bf16_t* out_w_bf = (bf16_t*)alloc((size_t)DD * DD * 2);
  bf16_t* w1_bf    = (bf16_t*)alloc((size_t)EE * DD * 2 * FFN2 * 2);
  bf16_t* w2_bf    = (bf16_t*)alloc((size_t)EE * FFN2 * DD * 2);
  bf16_t* xn_bf    = (bf16_t*)alloc((size_t)NT * DD * 2);
  float*  qkv_f    = (float*) alloc((size_t)NT * 3 * DD * 4);
  bf16_t* q_bf     = (bf16_t*)alloc((size_t)BB * NHH * TT * DHH * 2);
  bf16_t* k_bf     = (bf16_t*)alloc((size_t)BB * NHH * TT * DHH * 2);
  bf16_t* v_bf     = (bf16_t*)alloc((size_t)BB * NHH * TT * DHH * 2);
  bf16_t* o_bf     = (bf16_t*)alloc((size_t)NT * DD * 2);
  float*  xmid     = (float*) alloc((size_t)NT * DD * 4);
  bf16_t* xn2_bf   = (bf16_t*)alloc((size_t)NT * DD * 2);
  bf16_t* act_buf  = (bf16_t*)alloc((size_t)EE * CAP * FFN2 * 2);
  int*    lists    = (int*)   alloc((size_t)EE * CAP * 4);
  float*  gts      = (float*) alloc((size_t)EE * CAP * 4);
  int*    cnts     = (int*)   alloc(EE * 4);
  float*  pacc     = (float*) alloc(EE * 4);

  // weight precision conversion (one streaming pass)
  cvt_kernel<<<2048, 256, 0, stream>>>(qkv_w, qkv_w_bf, DD * 3 * DD);
  cvt_kernel<<<2048, 256, 0, stream>>>(out_w, out_w_bf, DD * DD);
  cvt_kernel<<<4096, 256, 0, stream>>>(w1, w1_bf, EE * DD * 2 * FFN2);
  cvt_kernel<<<4096, 256, 0, stream>>>(w2, w2_bf, EE * FFN2 * DD);

  // LN1 -> QKV -> RoPE -> attention -> out-proj(+residual, seeds d_out)
  ln_kernel<<<NT, 256, 0, stream>>>(x, ln1_g, ln1_b, xn_bf);
  gemm_bf16_kernel<<<dim3(NT / 128, 3 * DD / 64), 256, 0, stream>>>(
      xn_bf, DD, qkv_w_bf, 3 * DD, qkv_f, 3 * DD, DD, nullptr, nullptr);
  rope_kernel<<<(BB * TT * NHH * 32) / 256, 256, 0, stream>>>(qkv_f, q_bf, k_bf, v_bf);
  attn_kernel<<<BB * NHH * (TT / 16), 128, 0, stream>>>(q_bf, k_bf, v_bf, o_bf);
  gemm_bf16_kernel<<<dim3(NT / 128, DD / 64), 256, 0, stream>>>(
      o_bf, DD, out_w_bf, DD, xmid, DD, DD, x, out_x);

  // LN2 -> gate/route -> routed expert FFN -> weighted scatter -> loss
  ln_kernel<<<NT, 256, 0, stream>>>(xmid, ln2_g, ln2_b, xn2_bf);
  zero_kernel<<<1, 32, 0, stream>>>(cnts, pacc);
  gate_kernel<<<NT, 32, 0, stream>>>(xn2_bf, gate_w, ebias, cnts, lists, gts, pacc);
  moe_gemm1_kernel<<<dim3(CAP / 128, FFN2 / 64, EE), 256, 0, stream>>>(
      xn2_bf, w1_bf, lists, cnts, act_buf);
  moe_gemm2_kernel<<<dim3(CAP / 128, DD / 64, EE), 256, 0, stream>>>(
      act_buf, w2_bf, lists, cnts, gts, out_x);
  loss_kernel<<<1, 32, 0, stream>>>(cnts, pacc, out_loss);
}
